// MoELayer_42477226557897
// MI455X (gfx1250) — matrix operand order repaired and checked by reading, not compiled, not run on hardware
//
#include <hip/hip_runtime.h>
#include <hip/hip_bf16.h>
#include <math.h>

// Problem constants
#define N_TOK  4096            // B*T
#define D_DIM  1024
#define H_DIM  4096
#define E_NUM  8
// GEMM tiling
#define BM 64
#define BN 128
#define BK 64
#define LDA 72                 // padded bf16 row stride for A tile
#define LDB 72                 // padded bf16 row stride for B tile (stored [N][K])

typedef __attribute__((ext_vector_type(16))) __bf16 v16bf;
typedef __attribute__((ext_vector_type(8)))  __bf16 v8bf;
typedef __attribute__((ext_vector_type(4)))  __bf16 v4bf;
typedef __attribute__((ext_vector_type(8)))  float  v8f;

#define WMMA_BF16(A, B, C) \
  __builtin_amdgcn_wmma_f32_16x16x32_bf16(false, (A), false, (B), (short)0, \
                                          (C), false, false)

// ---------------------------------------------------------------------------
// fp32 [R][C] -> bf16 [C][R] transpose-convert (per expert), LDS tiled.
// Makes GEMM B-staging fully contiguous (global 16B loads -> LDS 16B stores)
// and makes the weight stream L2-resident (128 MB bf16 < 192 MB L2).
// ---------------------------------------------------------------------------
__global__ __launch_bounds__(256) void transpose_convert_kernel(
    const float* __restrict__ src, __bf16* __restrict__ dst, int R, int C) {
  __shared__ float tile[64][65];
  int e = blockIdx.z;
  int c0 = blockIdx.x * 64;
  int r0 = blockIdx.y * 64;
  const float* s = src + (size_t)e * R * C;
  __bf16* d = dst + (size_t)e * R * C;
#pragma unroll
  for (int j = 0; j < 4; ++j) {
    int idx = threadIdx.x + j * 256;   // 1024 float4 slots
    int r = idx >> 4, c4 = idx & 15;
    float4 v = *(const float4*)(s + (size_t)(r0 + r) * C + c0 + c4 * 4);
    tile[r][c4 * 4 + 0] = v.x; tile[r][c4 * 4 + 1] = v.y;
    tile[r][c4 * 4 + 2] = v.z; tile[r][c4 * 4 + 3] = v.w;
  }
  __syncthreads();
#pragma unroll
  for (int j = 0; j < 4; ++j) {
    int idx = threadIdx.x + j * 256;
    int c = idx >> 4, r4 = idx & 15;
    v4bf o = { (__bf16)tile[r4 * 4 + 0][c], (__bf16)tile[r4 * 4 + 1][c],
               (__bf16)tile[r4 * 4 + 2][c], (__bf16)tile[r4 * 4 + 3][c] };
    *(v4bf*)(d + (size_t)(c0 + c) * R + r0 + r4 * 4) = o;
  }
}

// ---------------------------------------------------------------------------
// Gating: logits, top-2, softmax, build per-expert pair lists
// ---------------------------------------------------------------------------
__global__ __launch_bounds__(128) void gating_kernel(
    const float* __restrict__ x, const float* __restrict__ gate_w,
    const float* __restrict__ gate_b, float* __restrict__ logits_out,
    int* __restrict__ cnt, int* __restrict__ list,
    float* __restrict__ pair_w, int* __restrict__ pair_e) {
  __shared__ float gws[D_DIM * E_NUM];   // 32 KB
  int tid = threadIdx.x;
  for (int i = tid; i < D_DIM * E_NUM; i += 128) gws[i] = gate_w[i];
  __syncthreads();

  int t = blockIdx.x * 128 + tid;
  float acc[E_NUM];
#pragma unroll
  for (int e = 0; e < E_NUM; ++e) acc[e] = gate_b[e];

  const float4* xr = (const float4*)(x + (size_t)t * D_DIM);
  for (int d4 = 0; d4 < D_DIM / 4; ++d4) {
    float4 xv = xr[d4];
    int db = d4 * 4;
#pragma unroll
    for (int e = 0; e < E_NUM; ++e) {
      acc[e] += xv.x * gws[(db + 0) * E_NUM + e] +
                xv.y * gws[(db + 1) * E_NUM + e] +
                xv.z * gws[(db + 2) * E_NUM + e] +
                xv.w * gws[(db + 3) * E_NUM + e];
    }
  }
#pragma unroll
  for (int e = 0; e < E_NUM; ++e) logits_out[t * E_NUM + e] = acc[e];

  int i0 = 0; float m0v = acc[0];
#pragma unroll
  for (int e = 1; e < E_NUM; ++e) if (acc[e] > m0v) { m0v = acc[e]; i0 = e; }
  int i1 = (i0 == 0) ? 1 : 0; float m1v = acc[i1];
#pragma unroll
  for (int e = 0; e < E_NUM; ++e)
    if (e != i0 && acc[e] > m1v) { m1v = acc[e]; i1 = e; }

  float e1 = __expf(m1v - m0v);
  float den = 1.0f / (1.0f + e1);

  int s0 = t * 2, s1 = s0 + 1;
  pair_w[s0] = den;       pair_e[s0] = i0;
  pair_w[s1] = e1 * den;  pair_e[s1] = i1;
  int p0 = atomicAdd(&cnt[i0], 1); list[i0 * N_TOK + p0] = s0;
  int p1 = atomicAdd(&cnt[i1], 1); list[i1 * N_TOK + p1] = s1;
}

// ---------------------------------------------------------------------------
// WMMA fragment helpers (layouts per CDNA5 ISA 7.12.2, bf16 16x16x32)
// ---------------------------------------------------------------------------
__device__ __forceinline__ v16bf frag16(const __bf16* lo, const __bf16* hi) {
  v8bf a = *(const v8bf*)lo;
  v8bf b = *(const v8bf*)hi;
  return __builtin_shufflevector(a, b, 0, 1, 2, 3, 4, 5, 6, 7,
                                 8, 9, 10, 11, 12, 13, 14, 15);
}

__device__ __forceinline__ float gelu_exact(float v) {
  return 0.5f * v * (1.0f + erff(v * 0.70710678118654752f));
}

// ---------------------------------------------------------------------------
// GEMM1: h = gelu( (w * x_gathered) @ w1_e + b1_e ), h stored bf16.
// Block tile 64x128, 8 waves in 2(M)x4(N); each wave owns 2x2 16x16 C tiles.
// Double-buffered LDS: one barrier per K-slab; next slab's global loads are
// issued before the WMMA burst so memory latency hides under compute.
// ---------------------------------------------------------------------------
__global__ __launch_bounds__(256) void moe_gemm1_kernel(
    const float* __restrict__ x, const __bf16* __restrict__ w1t,
    const float* __restrict__ b1, const int* __restrict__ cnt,
    const int* __restrict__ list, const float* __restrict__ pair_w,
    __bf16* __restrict__ hbuf) {
  int e = blockIdx.z;
  int count = cnt[e];
  int m0 = blockIdx.x * BM;
  if (m0 >= count) return;
  int n0 = blockIdx.y * BN;

  __shared__ __bf16 As[2][BM * LDA];   //  9.2 KB x2
  __shared__ __bf16 Bs[2][BN * LDB];   // 18.4 KB x2

  int tid = threadIdx.x;
  int lane = tid & 31;
  int wave = tid >> 5;
  int waveM = wave & 1;
  int waveN = wave >> 1;
  const int* mylist = list + e * N_TOK;

  // ---- A staging metadata, invariant across the K loop ----
  const float4* a_src[4]; float a_w[4]; int a_off[4];
#pragma unroll
  for (int j = 0; j < 4; ++j) {
    int idx = tid + j * 256;           // 1024 float4 slots (64 rows x 16)
    int r = idx >> 4, k4 = idx & 15;
    int gm = m0 + r;
    int gi = gm < count ? gm : count - 1;
    int slot = mylist[gi];
    a_w[j] = (gm < count) ? pair_w[slot] : 0.0f;
    a_src[j] = (const float4*)(x + (size_t)(slot >> 1) * D_DIM) + k4;
    a_off[j] = r * LDA + k4 * 4;
  }
  // ---- B staging metadata (w1t layout: [e][h][d], contiguous in K=d) ----
  const v8bf* b_src[4]; int b_off[4];
  const __bf16* wbase = w1t + (size_t)e * D_DIM * H_DIM;
#pragma unroll
  for (int j = 0; j < 4; ++j) {
    int idx = tid + j * 256;           // 1024 v8bf slots (128 rows x 8)
    int n = idx >> 3, k8 = idx & 7;
    b_src[j] = (const v8bf*)(wbase + (size_t)(n0 + n) * D_DIM + k8 * 8);
    b_off[j] = n * LDB + k8 * 8;
  }

  v8f c00 = {}, c01 = {}, c10 = {}, c11 = {};
  int mrow = lane & 15;
  int kba = (lane >> 4) * 8;
  int kbb = (lane >> 4) * 16;
  int aoff = (waveM * 32 + mrow) * LDA;
  int boff = (waveN * 32 + mrow) * LDB;

  // Prologue: stage slab 0
  float4 ar[4]; v8bf br[4];
#pragma unroll
  for (int j = 0; j < 4; ++j) { ar[j] = a_src[j][0]; br[j] = b_src[j][0]; }
#pragma unroll
  for (int j = 0; j < 4; ++j) {
    v4bf o = { (__bf16)(a_w[j] * ar[j].x), (__bf16)(a_w[j] * ar[j].y),
               (__bf16)(a_w[j] * ar[j].z), (__bf16)(a_w[j] * ar[j].w) };
    *(v4bf*)&As[0][a_off[j]] = o;
    *(v8bf*)&Bs[0][b_off[j]] = br[j];
  }
  __syncthreads();

  const int NSLAB = D_DIM / BK;
  for (int s = 0; s < NSLAB; ++s) {
    int cur = s & 1;
    int k0n = (s + 1) * BK;
    if (s + 1 < NSLAB) {               // issue next slab's loads early
#pragma unroll
      for (int j = 0; j < 4; ++j) {
        ar[j] = a_src[j][k0n >> 2];
        br[j] = b_src[j][k0n >> 3];
      }
    }
    const __bf16* arow0 = &As[cur][aoff];
    const __bf16* arow1 = arow0 + 16 * LDA;
    const __bf16* brow0 = &Bs[cur][boff];
    const __bf16* brow1 = brow0 + 16 * LDB;
#pragma unroll
    for (int kk = 0; kk < BK; kk += 32) {
      v16bf a0 = frag16(arow0 + kk + kba, arow0 + kk + kba + 16);
      v16bf a1 = frag16(arow1 + kk + kba, arow1 + kk + kba + 16);
      v16bf b0 = frag16(brow0 + kk + kba, brow0 + kk + kba + 16);
      v16bf b1f = frag16(brow1 + kk + kba, brow1 + kk + kba + 16);
      c00 = WMMA_BF16(a0, b0, c00);
      c01 = WMMA_BF16(a0, b1f, c01);
      c10 = WMMA_BF16(a1, b0, c10);
      c11 = WMMA_BF16(a1, b1f, c11);
    }
    if (s + 1 < NSLAB) {               // fill the other buffer
      int nxt = cur ^ 1;
#pragma unroll
      for (int j = 0; j < 4; ++j) {
        v4bf o = { (__bf16)(a_w[j] * ar[j].x), (__bf16)(a_w[j] * ar[j].y),
                   (__bf16)(a_w[j] * ar[j].z), (__bf16)(a_w[j] * ar[j].w) };
        *(v4bf*)&As[nxt][a_off[j]] = o;
        *(v8bf*)&Bs[nxt][b_off[j]] = br[j];
      }
    }
    __syncthreads();
  }

  // Epilogue: + b1, exact GELU, scatter-store bf16 h per pair slot
  int colbase = lane & 15;
  int rowoff = (lane >> 4) * 8;
  int h0c = n0 + waveN * 32 + colbase;
  const float* b1e = b1 + e * H_DIM;
  float bb0 = b1e[h0c], bb1 = b1e[h0c + 16];
#pragma unroll
  for (int r = 0; r < 8; ++r) {
    int rt0 = m0 + waveM * 32 + rowoff + r;
    int rt1 = rt0 + 16;
    if (rt0 < count) {
      int slot = mylist[rt0];
      __bf16* hp = hbuf + (size_t)slot * H_DIM;
      hp[h0c]      = (__bf16)gelu_exact(c00[r] + bb0);
      hp[h0c + 16] = (__bf16)gelu_exact(c01[r] + bb1);
    }
    if (rt1 < count) {
      int slot = mylist[rt1];
      __bf16* hp = hbuf + (size_t)slot * H_DIM;
      hp[h0c]      = (__bf16)gelu_exact(c10[r] + bb0);
      hp[h0c + 16] = (__bf16)gelu_exact(c11[r] + bb1);
    }
  }
}

// ---------------------------------------------------------------------------
// GEMM2: pair_out = h @ w2_e   (b2 and gate weight applied in combine)
// ---------------------------------------------------------------------------
__global__ __launch_bounds__(256) void moe_gemm2_kernel(
    const __bf16* __restrict__ hbuf, const __bf16* __restrict__ w2t,
    const int* __restrict__ cnt, const int* __restrict__ list,
    float* __restrict__ pair_out) {
  int e = blockIdx.z;
  int count = cnt[e];
  int m0 = blockIdx.x * BM;
  if (m0 >= count) return;
  int n0 = blockIdx.y * BN;

  __shared__ __bf16 As[2][BM * LDA];
  __shared__ __bf16 Bs[2][BN * LDB];

  int tid = threadIdx.x;
  int lane = tid & 31;
  int wave = tid >> 5;
  int waveM = wave & 1;
  int waveN = wave >> 1;
  const int* mylist = list + e * N_TOK;

  // ---- A staging metadata (h rows, already bf16) ----
  const v8bf* a_src[2]; int a_off[2]; bool a_val[2];
#pragma unroll
  for (int j = 0; j < 2; ++j) {
    int idx = tid + j * 256;           // 512 v8bf slots (64 rows x 8)
    int r = idx >> 3, k8 = idx & 7;
    int gm = m0 + r;
    int gi = gm < count ? gm : count - 1;
    int slot = mylist[gi];
    a_val[j] = gm < count;
    a_src[j] = (const v8bf*)(hbuf + (size_t)slot * H_DIM + k8 * 8);
    a_off[j] = r * LDA + k8 * 8;
  }
  // ---- B staging metadata (w2t layout: [e][d][h], contiguous in K=h) ----
  const v8bf* b_src[4]; int b_off[4];
  const __bf16* wbase = w2t + (size_t)e * H_DIM * D_DIM;
#pragma unroll
  for (int j = 0; j < 4; ++j) {
    int idx = tid + j * 256;
    int n = idx >> 3, k8 = idx & 7;
    b_src[j] = (const v8bf*)(wbase + (size_t)(n0 + n) * H_DIM + k8 * 8);
    b_off[j] = n * LDB + k8 * 8;
  }

  v8f c00 = {}, c01 = {}, c10 = {}, c11 = {};
  int mrow = lane & 15;
  int kba = (lane >> 4) * 8;
  int kbb = (lane >> 4) * 16;
  int aoff = (waveM * 32 + mrow) * LDA;
  int boff = (waveN * 32 + mrow) * LDB;

  v8bf zed = {};
  v8bf ar[2], br[4];
#pragma unroll
  for (int j = 0; j < 2; ++j) ar[j] = a_val[j] ? a_src[j][0] : zed;
#pragma unroll
  for (int j = 0; j < 4; ++j) br[j] = b_src[j][0];
#pragma unroll
  for (int j = 0; j < 2; ++j) *(v8bf*)&As[0][a_off[j]] = ar[j];
#pragma unroll
  for (int j = 0; j < 4; ++j) *(v8bf*)&Bs[0][b_off[j]] = br[j];
  __syncthreads();

  const int NSLAB = H_DIM / BK;
  for (int s = 0; s < NSLAB; ++s) {
    int cur = s & 1;
    int k0n = (s + 1) * BK;
    if (s + 1 < NSLAB) {
#pragma unroll
      for (int j = 0; j < 2; ++j)
        ar[j] = a_val[j] ? a_src[j][k0n >> 3] : zed;
#pragma unroll
      for (int j = 0; j < 4; ++j) br[j] = b_src[j][k0n >> 3];
    }
    const __bf16* arow0 = &As[cur][aoff];
    const __bf16* arow1 = arow0 + 16 * LDA;
    const __bf16* brow0 = &Bs[cur][boff];
    const __bf16* brow1 = brow0 + 16 * LDB;
#pragma unroll
    for (int kk = 0; kk < BK; kk += 32) {
      v16bf a0 = frag16(arow0 + kk + kba, arow0 + kk + kba + 16);
      v16bf a1 = frag16(arow1 + kk + kba, arow1 + kk + kba + 16);
      v16bf b0 = frag16(brow0 + kk + kba, brow0 + kk + kba + 16);
      v16bf b1f = frag16(brow1 + kk + kba, brow1 + kk + kba + 16);
      c00 = WMMA_BF16(a0, b0, c00);
      c01 = WMMA_BF16(a0, b1f, c01);
      c10 = WMMA_BF16(a1, b0, c10);
      c11 = WMMA_BF16(a1, b1f, c11);
    }
    if (s + 1 < NSLAB) {
      int nxt = cur ^ 1;
#pragma unroll
      for (int j = 0; j < 2; ++j) *(v8bf*)&As[nxt][a_off[j]] = ar[j];
#pragma unroll
      for (int j = 0; j < 4; ++j) *(v8bf*)&Bs[nxt][b_off[j]] = br[j];
    }
    __syncthreads();
  }

  int colbase = lane & 15;
  int rowoff = (lane >> 4) * 8;
  int dbase = n0 + waveN * 32 + colbase;
#pragma unroll
  for (int r = 0; r < 8; ++r) {
    int rt0 = m0 + waveM * 32 + rowoff + r;
    int rt1 = rt0 + 16;
    if (rt0 < count) {
      int slot = mylist[rt0];
      float* po = pair_out + (size_t)slot * D_DIM;
      po[dbase]      = c00[r];
      po[dbase + 16] = c01[r];
    }
    if (rt1 < count) {
      int slot = mylist[rt1];
      float* po = pair_out + (size_t)slot * D_DIM;
      po[dbase]      = c10[r];
      po[dbase + 16] = c11[r];
    }
  }
}

// ---------------------------------------------------------------------------
// Combine: out[tok,d] = sum_k w_k * (pair_out[tok*2+k, d] + b2[e_k, d])
// ---------------------------------------------------------------------------
__global__ __launch_bounds__(256) void combine_kernel(
    const float* __restrict__ pair_out, const float* __restrict__ pair_w,
    const int* __restrict__ pair_e, const float* __restrict__ b2,
    float* __restrict__ out) {
  int idx = blockIdx.x * 256 + threadIdx.x;   // over N_TOK * D_DIM
  int tok = idx >> 10;
  int d = idx & (D_DIM - 1);
  int s0 = tok * 2, s1 = s0 + 1;
  float v0 = pair_out[(size_t)s0 * D_DIM + d] + b2[pair_e[s0] * D_DIM + d];
  float v1 = pair_out[(size_t)s1 * D_DIM + d] + b2[pair_e[s1] * D_DIM + d];
  out[idx] = pair_w[s0] * v0 + pair_w[s1] * v1;
}

// ---------------------------------------------------------------------------
extern "C" void kernel_launch(void* const* d_in, const int* in_sizes, int n_in,
                              void* d_out, int out_size, void* d_ws,
                              size_t ws_size, hipStream_t stream) {
  (void)in_sizes; (void)n_in; (void)out_size; (void)ws_size;
  const float* x      = (const float*)d_in[0];
  const float* gate_w = (const float*)d_in[1];
  const float* gate_b = (const float*)d_in[2];
  const float* w1     = (const float*)d_in[3];
  const float* b1     = (const float*)d_in[4];
  const float* w2     = (const float*)d_in[5];
  const float* b2     = (const float*)d_in[6];
  float* out = (float*)d_out;
  float* logits_out = out + (size_t)N_TOK * D_DIM;   // outputs concatenated

  // Workspace carve-up (~235 MB)
  char* p = (char*)d_ws;
  int*    cnt      = (int*)p;    p += 256;
  int*    list     = (int*)p;    p += (size_t)E_NUM * N_TOK * 4;
  float*  pair_w   = (float*)p;  p += (size_t)N_TOK * 2 * 4;
  int*    pair_e   = (int*)p;    p += (size_t)N_TOK * 2 * 4;
  __bf16* w1t      = (__bf16*)p; p += (size_t)E_NUM * D_DIM * H_DIM * 2;
  __bf16* w2t      = (__bf16*)p; p += (size_t)E_NUM * H_DIM * D_DIM * 2;
  __bf16* hbuf     = (__bf16*)p; p += (size_t)N_TOK * 2 * H_DIM * 2;
  float*  pair_out = (float*)p;  p += (size_t)N_TOK * 2 * D_DIM * 4;

  hipMemsetAsync(cnt, 0, 256, stream);

  // w1: [E][D][H] fp32 -> w1t [E][H][D] bf16 ; w2: [E][H][D] -> w2t [E][D][H]
  transpose_convert_kernel<<<dim3(H_DIM / 64, D_DIM / 64, E_NUM), 256, 0,
                             stream>>>(w1, w1t, D_DIM, H_DIM);
  transpose_convert_kernel<<<dim3(D_DIM / 64, H_DIM / 64, E_NUM), 256, 0,
                             stream>>>(w2, w2t, H_DIM, D_DIM);

  gating_kernel<<<N_TOK / 128, 128, 0, stream>>>(x, gate_w, gate_b, logits_out,
                                                 cnt, list, pair_w, pair_e);

  moe_gemm1_kernel<<<dim3(N_TOK / BM, H_DIM / BN, E_NUM), 256, 0, stream>>>(
      x, w1t, b1, cnt, list, pair_w, hbuf);

  moe_gemm2_kernel<<<dim3(N_TOK / BM, D_DIM / BN, E_NUM), 256, 0, stream>>>(
      hbuf, w2t, cnt, list, pair_out);

  combine_kernel<<<(N_TOK * D_DIM) / 256, 256, 0, stream>>>(
      pair_out, pair_w, pair_e, b2, out);
}
